// RGAT_9689446220171
// MI455X (gfx1250) — matrix operand order repaired and checked by reading, not compiled, not run on hardware
//
#include <hip/hip_runtime.h>
#include <hip/hip_bf16.h>
#include <math.h>

// ---------------- problem constants (match reference) ----------------
#define NN    30000
#define DD    512
#define EE    300000
#define BB    4096
#define NRES  2
#define NDNN  2

typedef __attribute__((ext_vector_type(16))) __bf16 v16bf;
typedef __attribute__((ext_vector_type(8)))  __bf16 v8bf;
typedef __attribute__((ext_vector_type(8)))  float  v8f;

__device__ __forceinline__ __bf16 f2bf(float f) { return (__bf16)f; }

// =====================================================================
// GEMM: C[M,512] = A[M,512](f32) @ W[512,512] + bias, W given TRANSPOSED
// in bf16 (WT[n][k]).  Block: 256 thr = 8 waves (4x2), block tile 128x128,
// wave tile 32x64 -> 8 x v_wmma_f32_16x16x32_bf16 per wave per K-step.
// Ping-pong LDS (one barrier per K-step); A converted to bf16 at fetch
// time so staged state is only 16 VGPRs (no scratch spills).
// LDS layouts are fragment-ready:
//   As2[row][p] with K reordered [0..7,16..23,8..15,24..31]
//   BsT[n][p]   same K reorder (weights pre-transposed)
// so every fragment is one 32-byte contiguous LDS read (2x ds_load_b128).
// =====================================================================
#define BM 128
#define BN 128
#define BK 32
#define LDP (BK + 16)   // 48 bf16 = 96B row stride: keeps 16B/32B alignment

__global__ __launch_bounds__(256)
void gemm_bf16_kernel(const float* __restrict__ A, const __bf16* __restrict__ WT,
                      const float* __restrict__ bias, float* __restrict__ C, int M)
{
    __shared__ __bf16 As2[2][BM][LDP];
    __shared__ __bf16 BsT[2][BN][LDP];

    const int tid   = threadIdx.x;
    const int wave  = tid >> 5;
    const int lane  = tid & 31;
    const int waveM = wave >> 1;          // 0..3 -> 32-row strip
    const int waveN = wave & 1;           // 0..1 -> 64-col strip
    const int bm    = blockIdx.x * BM;
    const int bn    = blockIdx.y * BN;

    const int r16   = lane & 15;
    const int khalf = lane >> 4;

    // staging maps: each thread moves 16 elements of A and 16 of WT per K-step
    const int arow  = tid >> 1;               // 0..127
    const int acb   = (tid & 1) * 16;         // k chunk base: 0 or 16
    const int aposL = (acb == 0) ? 0 : 8;     // reordered LDS position of k[acb..acb+7]
    const int brow  = tid >> 1;               // 0..127 (n within tile)
    const int bkc   = (tid & 1) * 16;         // k base: 0 or 16

    // branch-free OOB handling: clamped rows only feed outputs that are
    // never stored (LDS row r -> output row bm+r), so clamping is safe.
    const int agr = (bm + arow < M) ? (bm + arow) : (M - 1);
    const float4* __restrict__ aptr = (const float4*)(A + (size_t)agr * DD + acb);
    const uint4*  __restrict__ bptr = (const uint4*)(WT + (size_t)(bn + brow) * DD + bkc);

    v8f acc[2][4];
    #pragma unroll
    for (int i = 0; i < 2; ++i)
        #pragma unroll
        for (int j = 0; j < 4; ++j)
            acc[i][j] = (v8f)(0.0f);

    v8bf  sa0, sa1;       // staged A (already bf16)
    uint4 rb0, rb1;       // staged B

    auto fetch = [&](int k0) {
        const float4* s = aptr + (k0 >> 2);     // k0 floats -> k0/4 float4
        const float4 t0 = s[0], t1 = s[1], t2 = s[2], t3 = s[3];
        sa0[0] = f2bf(t0.x); sa0[1] = f2bf(t0.y); sa0[2] = f2bf(t0.z); sa0[3] = f2bf(t0.w);
        sa0[4] = f2bf(t1.x); sa0[5] = f2bf(t1.y); sa0[6] = f2bf(t1.z); sa0[7] = f2bf(t1.w);
        sa1[0] = f2bf(t2.x); sa1[1] = f2bf(t2.y); sa1[2] = f2bf(t2.z); sa1[3] = f2bf(t2.w);
        sa1[4] = f2bf(t3.x); sa1[5] = f2bf(t3.y); sa1[6] = f2bf(t3.z); sa1[7] = f2bf(t3.w);
        const uint4* b = bptr + (k0 >> 3);      // k0 bf16 -> k0/8 uint4
        rb0 = b[0]; rb1 = b[1];
    };
    auto stage = [&](int buf) {
        *(v8bf*)&As2[buf][arow][aposL]      = sa0;   // k[acb..acb+7]
        *(v8bf*)&As2[buf][arow][aposL + 16] = sa1;   // k[acb+8..acb+15]
        *(uint4*)&BsT[buf][brow][bkc >> 1]        = rb0;
        *(uint4*)&BsT[buf][brow][(bkc >> 1) + 16] = rb1;
    };

    fetch(0);
    stage(0);
    __syncthreads();

    const int NIT = DD / BK;    // 16
    #pragma unroll 1
    for (int it = 0; it < NIT; ++it) {
        const int cur = it & 1;

        if (it + 1 < NIT) {
            fetch((it + 1) * BK);               // global loads fly during WMMA
            if (it + 2 < NIT) {                 // cache prefetch (global_prefetch_b8)
                __builtin_prefetch((const char*)(aptr + (((it + 2) * BK) >> 2)), 0, 1);
                __builtin_prefetch((const char*)(bptr + (((it + 2) * BK) >> 3)), 0, 1);
            }
        }

        // compute from current buffer: fragment-ready 32B contiguous LDS reads
        v16bf af[2];
        #pragma unroll
        for (int mi = 0; mi < 2; ++mi)
            af[mi] = *(const v16bf*)&As2[cur][waveM * 32 + mi * 16 + r16][khalf * 16];

        #pragma unroll
        for (int ni = 0; ni < 4; ++ni) {
            const v16bf bfv = *(const v16bf*)&BsT[cur][waveN * 64 + ni * 16 + r16][khalf * 16];
            #pragma unroll
            for (int mi = 0; mi < 2; ++mi) {
                acc[mi][ni] = __builtin_amdgcn_wmma_f32_16x16x32_bf16(
                    false, af[mi], false, bfv, (short)0, acc[mi][ni], false, false);
            }
        }

        if (it + 1 < NIT) stage(1 - cur);       // write other buffer: no WAR on cur
        __syncthreads();                        // single barrier per K-step
    }

    // store C (+bias): D layout: VGPR v -> row (lane>>4)*8+v, col lane&15
    #pragma unroll
    for (int mi = 0; mi < 2; ++mi) {
        #pragma unroll
        for (int ni = 0; ni < 4; ++ni) {
            const int col = bn + waveN * 64 + ni * 16 + r16;
            const float b = bias[col];
            #pragma unroll
            for (int v = 0; v < 8; ++v) {
                const int m = bm + waveM * 32 + mi * 16 + khalf * 8 + v;
                if (m < M) C[(size_t)m * DD + col] = acc[mi][ni][v] + b;
            }
        }
    }
}

// =====================================================================
// utility kernels
// =====================================================================
// bf16 convert with transpose: out[n][k] = in[k][n]  (per 512x512 matrix)
__global__ __launch_bounds__(256)
void cvt_bf16_t_kernel(const float* __restrict__ in, __bf16* __restrict__ out)
{
    const int mat = blockIdx.y;
    const float* src = in  + (size_t)mat * DD * DD;
    __bf16*      dst = out + (size_t)mat * DD * DD;
    for (int i = blockIdx.x * blockDim.x + threadIdx.x; i < DD * DD;
         i += gridDim.x * blockDim.x) {
        const int n = i >> 9, k = i & (DD - 1);
        dst[i] = f2bf(src[(size_t)k * DD + n]);
    }
}

__global__ __launch_bounds__(256)
void zero_kernel(float* __restrict__ p, int n)
{
    for (int i = blockIdx.x * blockDim.x + threadIdx.x; i < n; i += gridDim.x * blockDim.x)
        p[i] = 0.0f;
}

__global__ __launch_bounds__(256)
void copy_kernel(const float* __restrict__ a, float* __restrict__ b, int n)
{
    for (int i = blockIdx.x * blockDim.x + threadIdx.x; i < n; i += gridDim.x * blockDim.x)
        b[i] = a[i];
}

// per-node attention scores: sd = h . w_att[:D], ss = h . w_att[D:]
__global__ __launch_bounds__(256)
void att_scores_kernel(const float* __restrict__ h, const float* __restrict__ w_att,
                       float* __restrict__ sd, float* __restrict__ ss, int n)
{
    const int wave = threadIdx.x >> 5;
    const int lane = threadIdx.x & 31;
    const int node = blockIdx.x * 8 + wave;
    if (node >= n) return;
    const float* row = h + (size_t)node * DD;
    float ad = 0.f, as = 0.f;
    for (int d = lane; d < DD; d += 32) {
        const float v = row[d];
        ad += v * w_att[d];
        as += v * w_att[DD + d];
    }
    for (int o = 16; o > 0; o >>= 1) {
        ad += __shfl_xor(ad, o);
        as += __shfl_xor(as, o);
    }
    if (lane == 0) { sd[node] = ad; ss[node] = as; }
}

// edge pass 1: a = exp(tanh(sd[di]+ss[si]+b_att)); denom[di] += a
// (tanh output is in (-1,1) so exp() needs no max-shift for stability)
__global__ __launch_bounds__(256)
void edge_softmax_kernel(const int* __restrict__ edges, const float* __restrict__ sd,
                         const float* __restrict__ ss, const float* __restrict__ b_att,
                         float* __restrict__ aBuf, float* __restrict__ denom)
{
    const int j = blockIdx.x * blockDim.x + threadIdx.x;
    if (j >= EE) return;
    const int di = edges[j];
    const int si = edges[EE + j];
    const float e = tanhf(sd[di] + ss[si] + b_att[0]);
    const float a = expf(e);
    aBuf[j] = a;
    atomicAdd(&denom[di], a);
}

// edge pass 2: agg[di,:] += (a/denom[di]) * h[si,:]   (one block per edge)
__global__ __launch_bounds__(256)
void edge_agg_kernel(const int* __restrict__ edges, const float* __restrict__ aBuf,
                     const float* __restrict__ denom, const float* __restrict__ h,
                     float* __restrict__ agg)
{
    const int j  = blockIdx.x;
    const int di = edges[j];
    const int si = edges[EE + j];
    const float alpha = aBuf[j] / denom[di];
    const int d = threadIdx.x;
    const float* hr = h + (size_t)si * DD;
    float* ar = agg + (size_t)di * DD;
    atomicAdd(&ar[d],       alpha * hr[d]);
    atomicAdd(&ar[d + 256], alpha * hr[d + 256]);
}

// feat = 0.5 * (feat + agg + gat_bias)
__global__ __launch_bounds__(256)
void combine_kernel(float* __restrict__ feat, const float* __restrict__ agg,
                    const float* __restrict__ gat_bias, int n)
{
    for (int i = blockIdx.x * blockDim.x + threadIdx.x; i < n; i += gridDim.x * blockDim.x)
        feat[i] = 0.5f * (feat[i] + agg[i] + gat_bias[i & (DD - 1)]);
}

// gather rows: out[b,:] = feat[target[b],:]
__global__ __launch_bounds__(256)
void gather_kernel(const float* __restrict__ feat, const int* __restrict__ tgt,
                   float* __restrict__ out)
{
    const int b = blockIdx.x;
    const int t = tgt[b];
    const int d = threadIdx.x;
    out[(size_t)b * DD + d]       = feat[(size_t)t * DD + d];
    out[(size_t)b * DD + d + 256] = feat[(size_t)t * DD + d + 256];
}

// out = LayerNorm(tanh(x (+ resid))) * g + beta ; one block per row of 512
__global__ __launch_bounds__(256)
void tanh_ln_kernel(const float* __restrict__ x, const float* __restrict__ resid,
                    const float* __restrict__ g, const float* __restrict__ beta,
                    float* __restrict__ out)
{
    const int row = blockIdx.x;
    const int tid = threadIdx.x;
    const float* xr = x + (size_t)row * DD;
    float v0 = xr[tid], v1 = xr[tid + 256];
    if (resid) {
        v0 += resid[(size_t)row * DD + tid];
        v1 += resid[(size_t)row * DD + tid + 256];
    }
    v0 = tanhf(v0); v1 = tanhf(v1);
    float s  = v0 + v1;
    float sq = v0 * v0 + v1 * v1;
    for (int o = 16; o > 0; o >>= 1) {
        s  += __shfl_xor(s,  o);
        sq += __shfl_xor(sq, o);
    }
    __shared__ float sbuf[8], qbuf[8];
    const int wave = tid >> 5, lane = tid & 31;
    if (lane == 0) { sbuf[wave] = s; qbuf[wave] = sq; }
    __syncthreads();
    if (wave == 0) {
        float a = (lane < 8) ? sbuf[lane] : 0.f;
        float b = (lane < 8) ? qbuf[lane] : 0.f;
        for (int o = 4; o > 0; o >>= 1) { a += __shfl_xor(a, o); b += __shfl_xor(b, o); }
        if (lane == 0) { sbuf[0] = a; qbuf[0] = b; }
    }
    __syncthreads();
    const float mean = sbuf[0] * (1.0f / DD);
    const float var  = qbuf[0] * (1.0f / DD) - mean * mean;
    const float rstd = rsqrtf(var + 1e-5f);
    out[(size_t)row * DD + tid]       = (v0 - mean) * rstd * g[tid] + beta[tid];
    out[(size_t)row * DD + tid + 256] = (v1 - mean) * rstd * g[tid + 256] + beta[tid + 256];
}

// logits = sigmoid(h @ Wc + bc) ; one wave per row
__global__ __launch_bounds__(256)
void logits_kernel(const float* __restrict__ h, const float* __restrict__ Wc,
                   const float* __restrict__ bc, float* __restrict__ out)
{
    const int wave = threadIdx.x >> 5;
    const int lane = threadIdx.x & 31;
    const int row  = blockIdx.x * 8 + wave;
    if (row >= BB) return;
    const float* hr = h + (size_t)row * DD;
    float acc = 0.f;
    for (int d = lane; d < DD; d += 32) acc += hr[d] * Wc[d];
    for (int o = 16; o > 0; o >>= 1) acc += __shfl_xor(acc, o);
    if (lane == 0) out[row] = 1.0f / (1.0f + expf(-(acc + bc[0])));
}

// =====================================================================
// host orchestration
// =====================================================================
static inline int cdiv(int a, int b) { return (a + b - 1) / b; }

extern "C" void kernel_launch(void* const* d_in, const int* in_sizes, int n_in,
                              void* d_out, int out_size, void* d_ws, size_t ws_size,
                              hipStream_t stream)
{
    (void)in_sizes; (void)n_in; (void)out_size; (void)ws_size;

    const float* node_feat = (const float*)d_in[0];
    const int*   e_hop0    = (const int*)  d_in[1];
    const int*   e_hop1    = (const int*)  d_in[2];
    const int*   tgt       = (const int*)  d_in[3];
    const float* Wt        = (const float*)d_in[4];
    const float* bt        = (const float*)d_in[5];
    const float* Wf        = (const float*)d_in[6];
    const float* bf        = (const float*)d_in[7];
    const float* w_att     = (const float*)d_in[8];
    const float* b_att     = (const float*)d_in[9];
    const float* gat_bias  = (const float*)d_in[10];
    const float* Wp        = (const float*)d_in[11];
    const float* bp        = (const float*)d_in[12];
    const float* dnn_W     = (const float*)d_in[13];
    const float* dnn_b     = (const float*)d_in[14];
    const float* dnn_g     = (const float*)d_in[15];
    const float* dnn_beta  = (const float*)d_in[16];
    const float* res_g     = (const float*)d_in[17];
    const float* res_beta  = (const float*)d_in[18];
    const float* Wc        = (const float*)d_in[19];
    const float* bc        = (const float*)d_in[20];

    // ---- workspace partition ----
    char* p = (char*)d_ws;
    auto take = [&](size_t bytes) -> void* {
        void* r = (void*)p;
        p += (bytes + 255) & ~(size_t)255;
        return r;
    };
    float*  feat   = (float*) take((size_t)NN * DD * 4);
    float*  hbuf   = (float*) take((size_t)NN * DD * 4);
    float*  agg    = (float*) take((size_t)NN * DD * 4);
    __bf16* WtT    = (__bf16*)take((size_t)DD * DD * 2);   // transposed bf16 weights
    __bf16* WfT    = (__bf16*)take((size_t)DD * DD * 2);
    __bf16* WpT    = (__bf16*)take((size_t)DD * DD * 2);
    __bf16* dnnWT  = (__bf16*)take((size_t)NRES * NDNN * DD * DD * 2);
    float*  sd     = (float*) take((size_t)NN * 4);
    float*  ss     = (float*) take((size_t)NN * 4);
    float*  denom  = (float*) take((size_t)NN * 4);
    float*  aBuf   = (float*) take((size_t)EE * 4);
    float*  hb0    = (float*) take((size_t)BB * DD * 4);
    float*  hb1    = (float*) take((size_t)BB * DD * 4);
    float*  hsc    = (float*) take((size_t)BB * DD * 4);

    // ---- weight conversion to transposed bf16 ----
    dim3 cvt_grid(cdiv(DD * DD, 256), 1);
    cvt_bf16_t_kernel<<<cvt_grid, 256, 0, stream>>>(Wt, WtT);
    cvt_bf16_t_kernel<<<cvt_grid, 256, 0, stream>>>(Wf, WfT);
    cvt_bf16_t_kernel<<<cvt_grid, 256, 0, stream>>>(Wp, WpT);
    dim3 cvt_grid4(cdiv(DD * DD, 256), NRES * NDNN);
    cvt_bf16_t_kernel<<<cvt_grid4, 256, 0, stream>>>(dnn_W, dnnWT);

    // ---- node transform: feat = node_feat @ Wt + bt ----
    dim3 gemm_grid_N(cdiv(NN, BM), DD / BN);
    gemm_bf16_kernel<<<gemm_grid_N, 256, 0, stream>>>(node_feat, WtT, bt, feat, NN);

    // ---- two GAT hops: hop1 then hop0 (reference order) ----
    for (int hop = 0; hop < 2; ++hop) {
        const int* edges = (hop == 0) ? e_hop1 : e_hop0;
        gemm_bf16_kernel<<<gemm_grid_N, 256, 0, stream>>>(feat, WfT, bf, hbuf, NN);
        att_scores_kernel<<<cdiv(NN, 8), 256, 0, stream>>>(hbuf, w_att, sd, ss, NN);
        zero_kernel<<<cdiv(NN, 256), 256, 0, stream>>>(denom, NN);
        zero_kernel<<<4096, 256, 0, stream>>>(agg, NN * DD);
        edge_softmax_kernel<<<cdiv(EE, 256), 256, 0, stream>>>(edges, sd, ss, b_att, aBuf, denom);
        edge_agg_kernel<<<EE, 256, 0, stream>>>(edges, aBuf, denom, hbuf, agg);
        combine_kernel<<<4096, 256, 0, stream>>>(feat, agg, gat_bias, NN * DD);
    }

    // ---- head ----
    gather_kernel<<<BB, 256, 0, stream>>>(feat, tgt, hb0);

    dim3 gemm_grid_B(cdiv(BB, BM), DD / BN);
    gemm_bf16_kernel<<<gemm_grid_B, 256, 0, stream>>>(hb0, WpT, bp, hb1, BB);

    for (int r = 0; r < NRES; ++r) {
        copy_kernel<<<cdiv(BB * DD, 256), 256, 0, stream>>>(hb1, hsc, BB * DD);
        for (int i = 0; i < NDNN; ++i) {
            const int li = r * NDNN + i;
            gemm_bf16_kernel<<<gemm_grid_B, 256, 0, stream>>>(
                hb1, dnnWT + (size_t)li * DD * DD, dnn_b + (size_t)li * DD, hb0, BB);
            tanh_ln_kernel<<<BB, 256, 0, stream>>>(
                hb0, nullptr, dnn_g + (size_t)li * DD, dnn_beta + (size_t)li * DD, hb1);
        }
        tanh_ln_kernel<<<BB, 256, 0, stream>>>(
            hb1, hsc, res_g + (size_t)r * DD, res_beta + (size_t)r * DD, hb1);
    }

    logits_kernel<<<BB / 8, 256, 0, stream>>>(hb1, Wc, bc, (float*)d_out);
}
